// Experts_1726576853152
// MI455X (gfx1250) — matrix operand order repaired and checked by reading, not compiled, not run on hardware
//
#include <hip/hip_runtime.h>

// ---------------------------------------------------------------------------
// MoE FFN for MI455X (gfx1250): routing-sparse, bf16 WMMA, fp32 accumulate.
//   T=8192 tokens, d_model=1024, d_ff=4096, E=8 experts, top_k=2
// ---------------------------------------------------------------------------

#define T_TOK   8192
#define DMODEL  1024
#define DFF     4096
#define N_EXP   8
#define TOPK    2
#define NPAIR   (T_TOK * TOPK)       // 16384 (token,k) pairs total
#define CAP     NPAIR                // worst-case per-expert list capacity

#define TILE_M    32                 // tokens per workgroup
#define FF_CHUNK  128                // d_ff slice per fused step
#define XS_LD     1048               // padded LDS stride (12-bank rotation/row)
#define HS_LD     136                // padded LDS stride ( 4-bank rotation/row)

typedef __attribute__((ext_vector_type(16))) __bf16 bf16x16;
typedef __attribute__((ext_vector_type(8)))  __bf16 bf16x8;
typedef __attribute__((ext_vector_type(8)))  float  f32x8;
typedef __attribute__((ext_vector_type(4)))  int    v4i;

#if defined(__has_builtin)
#if __has_builtin(__builtin_amdgcn_global_load_async_to_lds_b128)
#define HAVE_ASYNC_LDS 1
#endif
#endif

// -------- workspace layout (bytes) --------
static const size_t OFF_X16   = 0;                                   // 16 MB
static const size_t OFF_WI16  = OFF_X16  + (size_t)T_TOK * DMODEL * 2;
static const size_t OFF_WO16  = OFF_WI16 + (size_t)N_EXP * DFF * DMODEL * 2; // 64 MB
static const size_t OFF_CNT   = OFF_WO16 + (size_t)N_EXP * DMODEL * DFF * 2; // 64 MB
static const size_t OFF_LISTS = OFF_CNT  + 256;
static const size_t OFF_YBUF  = OFF_LISTS + (size_t)N_EXP * CAP * 4;         // 0.5 MB
// ybuf: NPAIR x DMODEL fp32 = 64 MB.  Total ws use ~209 MB.

__device__ __forceinline__ f32x8 zero8() {
  f32x8 z;
#pragma unroll
  for (int i = 0; i < 8; ++i) z[i] = 0.0f;
  return z;
}

// A fragment (16x32, bf16) from row-major src (rows = M, cols = K).
// ISA layout: lanes 0-15 hold M=lane, K [0..8)+[16..24); lanes 16-31 hold
// M=lane-16, K [8..16)+[24..32).  Each 8-element K run is one 16B load.
__device__ __forceinline__ bf16x16 load_a(const __bf16* base, int ld,
                                          int row0, int k0, int lhalf, int l16) {
  const __bf16* p = base + (size_t)(row0 + l16) * ld + k0 + lhalf * 8;
  bf16x16 r;
  *(bf16x8*)&r       = *(const bf16x8*)p;
  *((bf16x8*)&r + 1) = *(const bf16x8*)(p + 16);
  return r;
}

// B fragment (32x16, bf16) taken from row-major weight W[n][k] (i.e. B = W^T).
// ISA layout: lanes 0-15 hold N=lane, K [0..8)+[16..24); lanes 16-31 hold K [8..16)+[24..32).
__device__ __forceinline__ bf16x16 load_b(const __bf16* W, int ld,
                                          int n0, int k0, int lhalf, int l16) {
  const __bf16* p = W + (size_t)(n0 + l16) * ld + k0 + lhalf * 8;
  bf16x16 r;
  *(bf16x8*)&r       = *(const bf16x8*)p;
  *((bf16x8*)&r + 1) = *(const bf16x8*)(p + 16);
  return r;
}

__device__ __forceinline__ f32x8 wmma_bf16(bf16x16 a, bf16x16 b, f32x8 c) {
  return __builtin_amdgcn_wmma_f32_16x16x32_bf16(false, a, false, b,
                                                 (short)0, c, false, false);
}

// ---------------------------------------------------------------------------
__global__ void zero_counts_k(int* counts) {
  if (threadIdx.x < N_EXP) counts[threadIdx.x] = 0;
}

__global__ void cvt_bf16_k(const float* __restrict__ src,
                           __bf16* __restrict__ dst, int n) {
  int base = (blockIdx.x * blockDim.x + threadIdx.x) * 4;
  if (base + 3 < n) {
    float4 v = *(const float4*)(src + base);
    dst[base + 0] = (__bf16)v.x;
    dst[base + 1] = (__bf16)v.y;
    dst[base + 2] = (__bf16)v.z;
    dst[base + 3] = (__bf16)v.w;
  } else {
    for (int i = base; i < n; ++i) dst[i] = (__bf16)src[i];
  }
}

// One list entry per (token,k) pair; entry stores pair index p = t*TOPK + k.
__global__ void build_lists_k(const int* __restrict__ sel,   // [T,K,E] one-hot
                              int* __restrict__ counts,
                              int* __restrict__ lists) {
  int p = blockIdx.x * blockDim.x + threadIdx.x;
  if (p >= NPAIR) return;
  const int* row = sel + (size_t)p * N_EXP;
  int e = 0;
#pragma unroll
  for (int j = 0; j < N_EXP; ++j)
    if (row[j] != 0) e = j;
  int pos = atomicAdd(&counts[e], 1);
  lists[e * CAP + pos] = p;
}

// ---------------------------------------------------------------------------
// Fused per-expert FFN: y = relu(X_g . wi_e^T) . wo_e^T for this expert's
// gathered rows.  One block = 32 token-pairs x full d_model output, 8 waves.
// ---------------------------------------------------------------------------
__global__ __launch_bounds__(256) void ffn_expert_k(
    int e,
    const __bf16* __restrict__ X16,
    const __bf16* __restrict__ Wi16,
    const __bf16* __restrict__ Wo16,
    const int* __restrict__ counts,
    const int* __restrict__ lists,
    float* __restrict__ ybuf) {
  __shared__ __bf16 Xs[TILE_M * XS_LD];       // 32 x 1024 X tile      ~65.5 KB
  __shared__ __bf16 Hs[2][TILE_M * HS_LD];    // 32 x 128 h, dbl-buf   ~17 KB
  __shared__ int    Ps[TILE_M];               // pair index per row

  const int cnt  = counts[e];
  const int row0 = blockIdx.x * TILE_M;
  if (row0 >= cnt) return;                    // uniform block early-exit
  const int n_valid = min(TILE_M, cnt - row0);

  const __bf16* Wi_e   = Wi16 + (size_t)e * DFF * DMODEL;
  const __bf16* Wo_e   = Wo16 + (size_t)e * DMODEL * DFF;
  const int*    list_e = lists + e * CAP;

  const int tid = threadIdx.x;
  if (tid < TILE_M) {
    // padded rows reuse a valid pair index; their stores are masked later
    Ps[tid] = (tid < n_valid) ? list_e[row0 + tid] : list_e[row0];
  }
  __syncthreads();

  // gather 32 token rows of X (bf16) into LDS; 16B chunks, fully coalesced
#if HAVE_ASYNC_LDS
  for (int idx = tid; idx < TILE_M * (DMODEL / 8); idx += 256) {
    int r = idx >> 7;             // DMODEL/8 = 128 chunks per row
    int c = idx & 127;
    int t = Ps[r] >> 1;           // pair -> token
    __attribute__((address_space(1))) v4i* g =
        (__attribute__((address_space(1))) v4i*)(X16 + (size_t)t * DMODEL + c * 8);
    __attribute__((address_space(3))) v4i* l =
        (__attribute__((address_space(3))) v4i*)(Xs + r * XS_LD + c * 8);
    __builtin_amdgcn_global_load_async_to_lds_b128(g, l, 0, 0);
  }
#if __has_builtin(__builtin_amdgcn_s_wait_asynccnt)
  __builtin_amdgcn_s_wait_asynccnt(0);
#else
  asm volatile("s_wait_asynccnt 0x0" ::: "memory");
#endif
#else
  for (int idx = tid; idx < TILE_M * (DMODEL / 8); idx += 256) {
    int r = idx >> 7;
    int c = idx & 127;
    int t = Ps[r] >> 1;
    *(bf16x8*)(Xs + r * XS_LD + c * 8) =
        *(const bf16x8*)(X16 + (size_t)t * DMODEL + c * 8);
  }
#endif
  __syncthreads();

  const int wave  = tid >> 5;     // 8 wave32s
  const int lane  = tid & 31;
  const int lhalf = lane >> 4;
  const int l16   = lane & 15;

  const int ncol0 = wave * 128;   // phase-2 output column slice per wave

  f32x8 acc2[2][8];               // 32 x 128 fp32 out accumulators per wave
#pragma unroll
  for (int a = 0; a < 2; ++a)
#pragma unroll
    for (int b = 0; b < 8; ++b) acc2[a][b] = zero8();

  for (int c = 0; c < DFF / FF_CHUNK; ++c) {
    __bf16* HsW = Hs[c & 1];
    // ---- phase 1: h_chunk[32 x 128]; each wave owns a 32 x 16 f-slice
    //      (2 M-tiles share one B fragment -> 1.5 frag loads per WMMA)
    f32x8 acc1_0 = zero8();
    f32x8 acc1_1 = zero8();
    const int f0 = c * FF_CHUNK + wave * 16;
#pragma unroll 4
    for (int k = 0; k < DMODEL; k += 32) {
      bf16x16 b  = load_b(Wi_e, DMODEL, f0, k, lhalf, l16);
      bf16x16 a0 = load_a(Xs, XS_LD, 0,  k, lhalf, l16);
      bf16x16 a1 = load_a(Xs, XS_LD, 16, k, lhalf, l16);
      acc1_0 = wmma_bf16(a0, b, acc1_0);
      acc1_1 = wmma_bf16(a1, b, acc1_1);
    }
    // relu + cvt to bf16, scatter per C/D lane layout into the parity buffer.
    // Writing Hs[c&1] is safe: the barrier in chunk c-1 guarantees everyone
    // finished reading Hs[c&1] during chunk c-2.
#pragma unroll
    for (int r = 0; r < 8; ++r) {
      int   m0 = r + 8 * lhalf;
      float v0 = acc1_0[r];
      float v1 = acc1_1[r];
      v0 = v0 > 0.0f ? v0 : 0.0f;
      v1 = v1 > 0.0f ? v1 : 0.0f;
      HsW[m0 * HS_LD + wave * 16 + l16]        = (__bf16)v0;
      HsW[(m0 + 16) * HS_LD + wave * 16 + l16] = (__bf16)v1;
    }
    __syncthreads();              // one barrier per chunk (double-buffered)
    // ---- phase 2: out[32 x 128] += h_chunk . wo_chunk^T
#pragma unroll
    for (int kk = 0; kk < FF_CHUNK; kk += 32) {
      bf16x16 a0 = load_a(HsW, HS_LD, 0,  kk, lhalf, l16);
      bf16x16 a1 = load_a(HsW, HS_LD, 16, kk, lhalf, l16);
#pragma unroll
      for (int tn = 0; tn < 8; ++tn) {
        bf16x16 b = load_b(Wo_e, DFF, ncol0 + tn * 16,
                           c * FF_CHUNK + kk, lhalf, l16);
        acc2[0][tn] = wmma_bf16(a0, b, acc2[0][tn]);
        acc2[1][tn] = wmma_bf16(a1, b, acc2[1][tn]);
      }
    }
  }

  // epilogue: each pair row has a private ybuf row -> no atomics needed
#pragma unroll
  for (int tmi = 0; tmi < 2; ++tmi)
#pragma unroll
    for (int tn = 0; tn < 8; ++tn)
#pragma unroll
      for (int r = 0; r < 8; ++r) {
        int m = tmi * 16 + r + 8 * lhalf;
        if (m < n_valid) {
          size_t p = (size_t)Ps[m];
          ybuf[p * DMODEL + ncol0 + tn * 16 + l16] = acc2[tmi][tn][r];
        }
      }
}

// out[t,m] = rw[t,0]*y(t,0)[m] + rw[t,1]*y(t,1)[m]  (fixed order: deterministic)
__global__ void reduce_pairs_k(const float* __restrict__ rw,
                               const float* __restrict__ ybuf,
                               float* __restrict__ out) {
  int gid = blockIdx.x * blockDim.x + threadIdx.x;
  if (gid >= T_TOK * DMODEL) return;
  int t = gid >> 10;
  int m = gid & (DMODEL - 1);
  float y0 = ybuf[(size_t)(t * 2 + 0) * DMODEL + m];
  float y1 = ybuf[(size_t)(t * 2 + 1) * DMODEL + m];
  out[gid] = rw[t * 2 + 0] * y0 + rw[t * 2 + 1] * y1;
}

// ---------------------------------------------------------------------------
extern "C" void kernel_launch(void* const* d_in, const int* in_sizes, int n_in,
                              void* d_out, int out_size, void* d_ws, size_t ws_size,
                              hipStream_t stream) {
  const float* hs  = (const float*)d_in[0];   // [T, DMODEL] fp32
  const int*   sel = (const int*)d_in[1];     // [T, K, E] one-hot int32
  const float* rw  = (const float*)d_in[2];   // [T, K] fp32
  const float* wi  = (const float*)d_in[3];   // [E, DFF, DMODEL] fp32
  const float* wo  = (const float*)d_in[4];   // [E, DMODEL, DFF] fp32
  float* out = (float*)d_out;

  char* ws = (char*)d_ws;
  __bf16* X16   = (__bf16*)(ws + OFF_X16);
  __bf16* Wi16  = (__bf16*)(ws + OFF_WI16);
  __bf16* Wo16  = (__bf16*)(ws + OFF_WO16);
  int*    cnts  = (int*)(ws + OFF_CNT);
  int*    lists = (int*)(ws + OFF_LISTS);
  float*  ybuf  = (float*)(ws + OFF_YBUF);

  zero_counts_k<<<1, 32, 0, stream>>>(cnts);

  {
    int n = T_TOK * DMODEL;
    cvt_bf16_k<<<(n / 4 + 255) / 256, 256, 0, stream>>>(hs, X16, n);
  }
  {
    int n = N_EXP * DFF * DMODEL;
    cvt_bf16_k<<<(n / 4 + 255) / 256, 256, 0, stream>>>(wi, Wi16, n);
    cvt_bf16_k<<<(n / 4 + 255) / 256, 256, 0, stream>>>(wo, Wo16, n);
  }

  build_lists_k<<<(NPAIR + 255) / 256, 256, 0, stream>>>(sel, cnts, lists);

  // worst-case grid (all pairs on one expert); inactive blocks exit on count
  for (int e = 0; e < N_EXP; ++e) {
    ffn_expert_k<<<CAP / TILE_M, 256, 0, stream>>>(e, X16, Wi16, Wo16,
                                                   cnts, lists, ybuf);
  }

  reduce_pairs_k<<<(T_TOK * DMODEL + 255) / 256, 256, 0, stream>>>(rw, ybuf, out);
}
